// LogoCapsNet_8461085573247
// MI455X (gfx1250) — compile-verified, matrix operand order checked by reading, not run on hardware
//
#include <hip/hip_runtime.h>
#include <hip/hip_bf16.h>
#include <math.h>

typedef __attribute__((ext_vector_type(16))) _Float16 v16h;
typedef __attribute__((ext_vector_type(8)))  _Float16 v8h;
typedef __attribute__((ext_vector_type(8)))  float    v8f;

// ---------------- problem constants ----------------
#define BATCH   128
#define M1      (BATCH * 576)     // 73728 conv1 output pixels
#define K1PAD   256               // 243 padded to 256
#define M2      (BATCH * 64)      // 8192 conv2 output pixels
#define K2      20736             // 256*81, exactly 648*32
#define KSPLIT2 10368             // K2/2 = 324 k-steps of 32
#define TILE_H  (128 * 32)        // one LDS tile: 128 rows x 32 halfs
#define XIMG    4096              // padded f16 image: 3072 data + 1024 zeros

// ---------------------------------------------------------------------------
// WMMA fragment loader: LDS tile rows are K-contiguous (32 halfs = 64 B/row).
// A-operand 16x32 f16 layout (ISA 7.12.2): lanes 0-15 hold K 0-7 & 16-23,
// lanes 16-31 hold K 8-15 & 24-31  ->  two ds_load_b128 per fragment.
// B operand stored N-major (rows of K) uses the identical pattern.
// ---------------------------------------------------------------------------
__device__ __forceinline__ v16h frag_ld(const _Float16* base, int row0, int lane) {
  const int half = lane >> 4;
  const _Float16* p = base + (row0 + (lane & 15)) * 32;
  v8h lo = *(const v8h*)(p + half * 8);
  v8h hi = *(const v8h*)(p + 16 + half * 8);
  v16h f;
#pragma unroll
  for (int i = 0; i < 8; ++i) { f[i] = lo[i]; f[i + 8] = hi[i]; }
  return f;
}

struct Tile { v8h a0, a1, b0, b1; };

__device__ __forceinline__ void tile_store(_Float16* lA, _Float16* lB,
                                           int arow, int akk0, const Tile& t) {
  *(v8h*)&lA[arow * 32 + akk0]     = t.a0;
  *(v8h*)&lA[arow * 32 + akk0 + 8] = t.a1;
  *(v8h*)&lB[arow * 32 + akk0]     = t.b0;
  *(v8h*)&lB[arow * 32 + akk0 + 8] = t.b1;
}

__device__ __forceinline__ void tile_mma(const _Float16* lA, const _Float16* lB,
                                         int wm, int wn, int lane, v8f acc[4][2]) {
  v16h af[4], bf[2];
#pragma unroll
  for (int mi = 0; mi < 4; ++mi) af[mi] = frag_ld(lA, wm + mi * 16, lane);
#pragma unroll
  for (int ni = 0; ni < 2; ++ni) bf[ni] = frag_ld(lB, wn + ni * 16, lane);
#pragma unroll
  for (int mi = 0; mi < 4; ++mi)
#pragma unroll
    for (int ni = 0; ni < 2; ++ni)
      acc[mi][ni] = __builtin_amdgcn_wmma_f32_16x16x32_f16(
          false, af[mi], false, bf[ni], (short)0, acc[mi][ni], false, false);
}

// ---------------------------------------------------------------------------
// Branch-free register staging of one (A,B) chunk: 16 independent f16 gathers
// (compiler can clause them under one wait) + 2x b128 weight loads.
// ---------------------------------------------------------------------------
__device__ __forceinline__ Tile stage(const _Float16* __restrict__ src, int base,
                                      const int* __restrict__ koff,
                                      const _Float16* __restrict__ brow, int kk) {
  Tile t;
  int4 o0 = *(const int4*)(koff + kk);
  int4 o1 = *(const int4*)(koff + kk + 4);
  int4 o2 = *(const int4*)(koff + kk + 8);
  int4 o3 = *(const int4*)(koff + kk + 12);
  union { v8h v; _Float16 e[8]; } A0, A1;
  A0.e[0] = src[base + o0.x]; A0.e[1] = src[base + o0.y];
  A0.e[2] = src[base + o0.z]; A0.e[3] = src[base + o0.w];
  A0.e[4] = src[base + o1.x]; A0.e[5] = src[base + o1.y];
  A0.e[6] = src[base + o1.z]; A0.e[7] = src[base + o1.w];
  A1.e[0] = src[base + o2.x]; A1.e[1] = src[base + o2.y];
  A1.e[2] = src[base + o2.z]; A1.e[3] = src[base + o2.w];
  A1.e[4] = src[base + o3.x]; A1.e[5] = src[base + o3.y];
  A1.e[6] = src[base + o3.z]; A1.e[7] = src[base + o3.w];
  t.a0 = A0.v; t.a1 = A1.v;
  t.b0 = *(const v8h*)(brow + kk);
  t.b1 = *(const v8h*)(brow + kk + 8);
  return t;
}

// ---------------- precomputed im2col offset tables (no /81, /9 in hot loop) -
__global__ void __launch_bounds__(256) build_koff1(int* __restrict__ koff) {
  int k = threadIdx.x;                           // 256 entries
  if (k < 243) {
    int c = k / 81, r = k - c * 81, ky = r / 9, kx = r - ky * 9;
    koff[k] = c * 1024 + ky * 32 + kx;           // offset into padded xh image
  } else {
    koff[k] = 3072;                              // K pad -> zero tail of image
  }
}

__global__ void __launch_bounds__(256) build_koff2(int* __restrict__ koff) {
  int k = blockIdx.x * 256 + threadIdx.x;        // 20736 entries (81 blocks)
  int ic = k / 81, r = k - ic * 81, ky = r / 9, kx = r - ky * 9;
  koff[k] = ic * 576 + ky * 24 + kx;             // offset into h[bi] volume
}

// ---------------- input / weight conversion ----------------
// x -> padded f16 images: [bi][4096] = 3072 data halfs + 1024 zeros
__global__ void __launch_bounds__(256) cvt_x(const float* __restrict__ x,
                                             _Float16* __restrict__ xh) {
  int idx = blockIdx.x * 256 + threadIdx.x;      // 524288 = 128 * 4096
  int img = idx >> 12, o = idx & 4095;
  xh[idx] = (o < 3072) ? (_Float16)x[img * 3072 + o] : (_Float16)0.f;
}

__global__ void __launch_bounds__(256) cvt_w1(const float* __restrict__ w,
                                              _Float16* __restrict__ wt1) {
  int idx = blockIdx.x * 256 + threadIdx.x;      // 65536 = 256 oc * 256 kpad
  int n = idx >> 8, k = idx & 255;
  wt1[idx] = (k < 243) ? (_Float16)w[n * 243 + k] : (_Float16)0.f;
}

__global__ void __launch_bounds__(256) cvt_w2(const float* __restrict__ w,
                                              _Float16* __restrict__ wt2) {
  int idx = blockIdx.x * 256 + threadIdx.x;      // 5308416 = 256 oc * 20736
  wt2[idx] = (_Float16)w[idx];
}

__global__ void __launch_bounds__(256) zero_u(float4* __restrict__ u) {
  int idx = blockIdx.x * 256 + threadIdx.x;      // 524288 float4 = 2M floats
  u[idx] = make_float4(0.f, 0.f, 0.f, 0.f);
}

// ---------------------------------------------------------------------------
// Conv1 implicit GEMM: M=73728, N=256, K=256(pad). 128x128 tile, BK=32,
// LDS ping-pong (one barrier per k-step). bias + leaky_relu -> f16 h.
// ---------------------------------------------------------------------------
__global__ void __launch_bounds__(256)
conv1_wmma(const _Float16* __restrict__ xh, const _Float16* __restrict__ wt1,
           const int* __restrict__ koff, const float* __restrict__ cb,
           _Float16* __restrict__ h) {
  __shared__ __align__(16) _Float16 lA[2 * TILE_H];
  __shared__ __align__(16) _Float16 lB[2 * TILE_H];
  const int tid = threadIdx.x;
  const int lane = tid & 31, wv = tid >> 5;
  const int wm = (wv & 1) * 64, wn = (wv >> 1) * 32;
  const int m0 = blockIdx.x * 128, n0 = blockIdx.y * 128;

  const int arow = tid >> 1;
  const int akk0 = (tid & 1) * 16;
  const int gm = m0 + arow;
  const int bi = gm / 576, pix = gm - bi * 576;
  const int oy = pix / 24, ox = pix - oy * 24;
  const int xbase = bi * XIMG + oy * 32 + ox;
  const _Float16* brow = wt1 + (n0 + arow) * 256;

  v8f acc[4][2];
#pragma unroll
  for (int i = 0; i < 4; ++i)
#pragma unroll
    for (int j = 0; j < 2; ++j)
#pragma unroll
      for (int e = 0; e < 8; ++e) acc[i][j][e] = 0.f;

  Tile t = stage(xh, xbase, koff, brow, akk0);
  tile_store(lA, lB, arow, akk0, t);
  __syncthreads();
  int p = 0;
  for (int k0 = 0; k0 < K1PAD; k0 += 32) {
    if (k0 + 32 < K1PAD) {
      Tile nt = stage(xh, xbase, koff, brow, k0 + 32 + akk0);
      tile_store(lA + (p ^ 1) * TILE_H, lB + (p ^ 1) * TILE_H, arow, akk0, nt);
    }
    tile_mma(lA + p * TILE_H, lB + p * TILE_H, wm, wn, lane, acc);
    __syncthreads();
    p ^= 1;
  }

  // D layout: VGPR r -> M = r + 8*(lane>>4), N = lane&15
#pragma unroll
  for (int mi = 0; mi < 4; ++mi)
#pragma unroll
    for (int ni = 0; ni < 2; ++ni) {
      const int gn = n0 + wn + ni * 16 + (lane & 15);
      const float bias = cb[gn];
#pragma unroll
      for (int r = 0; r < 8; ++r) {
        const int gmr = m0 + wm + mi * 16 + r + ((lane >> 4) << 3);
        float vv = acc[mi][ni][r] + bias;
        vv = vv > 0.f ? vv : 0.01f * vv;             // leaky_relu
        const int bb = gmr / 576, pp = gmr - bb * 576;
        h[(bb * 256 + gn) * 576 + pp] = (_Float16)vv;
      }
    }
}

// ---------------------------------------------------------------------------
// Primary-caps conv implicit GEMM: M=8192, N=256, K=20736 (2-way K split via
// blockIdx.z; exactly two commutative f32 atomic contributions per element).
// ---------------------------------------------------------------------------
__global__ void __launch_bounds__(256)
conv2_wmma(const _Float16* __restrict__ h, const _Float16* __restrict__ wt2,
           const int* __restrict__ koff, const float* __restrict__ pb,
           float* __restrict__ u) {
  __shared__ __align__(16) _Float16 lA[2 * TILE_H];
  __shared__ __align__(16) _Float16 lB[2 * TILE_H];
  const int tid = threadIdx.x;
  const int lane = tid & 31, wv = tid >> 5;
  const int wm = (wv & 1) * 64, wn = (wv >> 1) * 32;
  const int m0 = blockIdx.x * 128, n0 = blockIdx.y * 128;
  const int kbeg = blockIdx.z * KSPLIT2, kend = kbeg + KSPLIT2;

  const int arow = tid >> 1;
  const int akk0 = (tid & 1) * 16;
  const int gm = m0 + arow;
  const int bi = gm >> 6, hw = gm & 63;
  const int hbase = bi * 147456 + (hw >> 3) * 48 + (hw & 7) * 2; // iy0*24+ix0
  const _Float16* brow = wt2 + (size_t)(n0 + arow) * K2;

  v8f acc[4][2];
#pragma unroll
  for (int i = 0; i < 4; ++i)
#pragma unroll
    for (int j = 0; j < 2; ++j)
#pragma unroll
      for (int e = 0; e < 8; ++e) acc[i][j][e] = 0.f;

  Tile t = stage(h, hbase, koff, brow, kbeg + akk0);
  tile_store(lA, lB, arow, akk0, t);
  __syncthreads();
  int p = 0;
  for (int k0 = kbeg; k0 < kend; k0 += 32) {
    if (k0 + 32 < kend) {
      Tile nt = stage(h, hbase, koff, brow, k0 + 32 + akk0);
      tile_store(lA + (p ^ 1) * TILE_H, lB + (p ^ 1) * TILE_H, arow, akk0, nt);
      if (k0 + 64 < kend)
        __builtin_prefetch(brow + k0 + 64 + akk0, 0, 0);  // global_prefetch
    }
    tile_mma(lA + p * TILE_H, lB + p * TILE_H, wm, wn, lane, acc);
    __syncthreads();
    p ^= 1;
  }

#pragma unroll
  for (int mi = 0; mi < 4; ++mi)
#pragma unroll
    for (int ni = 0; ni < 2; ++ni) {
      const int gn = n0 + wn + ni * 16 + (lane & 15);    // out ch = cap*8+i
      const float bias = (blockIdx.z == 0) ? pb[gn] : 0.f;
#pragma unroll
      for (int r = 0; r < 8; ++r) {
        const int gmr = m0 + wm + mi * 16 + r + ((lane >> 4) << 3);
        const int bb = gmr >> 6, hw2 = gmr & 63;
        const int node = hw2 * 32 + (gn >> 3);           // torch flatten order
        atomicAdd(u + ((size_t)bb * 2048 + node) * 8 + (gn & 7),
                  acc[mi][ni][r] + bias);
      }
    }
}

// ---------------- squash over the 8-vector of each node ----------------
__global__ void __launch_bounds__(256) squash_u(float* __restrict__ u) {
  int n = blockIdx.x * 256 + threadIdx.x;        // 262144 nodes
  float4* p = (float4*)(u + (size_t)n * 8);
  float4 a = p[0], b = p[1];
  float s = a.x * a.x + a.y * a.y + a.z * a.z + a.w * a.w +
            b.x * b.x + b.y * b.y + b.z * b.z + b.w * b.w;
  float sc = sqrtf(s) / (1.f + s);
  a.x *= sc; a.y *= sc; a.z *= sc; a.w *= sc;
  b.x *= sc; b.y *= sc; b.z *= sc; b.w *= sc;
  p[0] = a; p[1] = b;
}

// ---------------- dynamic routing: one block per batch element ----------------
#define RT 512
__device__ __forceinline__ void uhat_row(const float* __restrict__ uS,
                                         const float* __restrict__ W,
                                         int n, float uh[16]) {
#pragma unroll
  for (int d = 0; d < 16; ++d) uh[d] = 0.f;
#pragma unroll
  for (int i = 0; i < 8; ++i) {
    float ui = uS[n * 8 + i];
    const float4* w4 = (const float4*)(W + i * 16);
    float4 w0 = w4[0], w1 = w4[1], w2 = w4[2], w3 = w4[3];
    uh[0]  += ui * w0.x; uh[1]  += ui * w0.y; uh[2]  += ui * w0.z; uh[3]  += ui * w0.w;
    uh[4]  += ui * w1.x; uh[5]  += ui * w1.y; uh[6]  += ui * w1.z; uh[7]  += ui * w1.w;
    uh[8]  += ui * w2.x; uh[9]  += ui * w2.y; uh[10] += ui * w2.z; uh[11] += ui * w2.w;
    uh[12] += ui * w3.x; uh[13] += ui * w3.y; uh[14] += ui * w3.z; uh[15] += ui * w3.w;
  }
}

__global__ void __launch_bounds__(RT)
routing_kernel(const float* __restrict__ u, const float* __restrict__ dw,
               float* __restrict__ out) {
  extern __shared__ float sm[];
  float* uS   = sm;               // 2048*8   = 16384 f
  float* L    = uS + 2048 * 8;    // 10*2048  = 20480 f (logits)
  float* mx   = L + 10 * 2048;    // 2048 f
  float* dn   = mx + 2048;        // 2048 f
  float* red  = dn + 2048;        // RT*16    = 8192 f
  float* outS = red + RT * 16;    // 10*16 (squashed)
  const int b = blockIdx.x, tid = threadIdx.x;

  for (int i = tid; i < 2048 * 8 / 4; i += RT)
    ((float4*)uS)[i] = ((const float4*)(u + (size_t)b * 2048 * 8))[i];
  for (int i = tid; i < 10 * 2048; i += RT) L[i] = 0.f;
  __syncthreads();

  for (int it = 0; it < 4; ++it) {
    // softmax stats over categories, per node
    for (int n = tid; n < 2048; n += RT) {
      float m = L[n];
#pragma unroll
      for (int c = 1; c < 10; ++c) m = fmaxf(m, L[c * 2048 + n]);
      float s = 0.f;
#pragma unroll
      for (int c = 0; c < 10; ++c) s += __expf(L[c * 2048 + n] - m);
      mx[n] = m; dn[n] = s;
    }
    __syncthreads();

    // out[c] = squash( sum_n prob(c,n) * uhat(c,n,:) ), uhat streamed from L2
    for (int c = 0; c < 10; ++c) {
      float acc[16];
#pragma unroll
      for (int d = 0; d < 16; ++d) acc[d] = 0.f;
      for (int n = tid; n < 2048; n += RT) {
        float uh[16];
        uhat_row(uS, dw + (size_t)(c * 2048 + n) * 128, n, uh);
        float p = __expf(L[c * 2048 + n] - mx[n]) / dn[n];
#pragma unroll
        for (int d = 0; d < 16; ++d) acc[d] += uh[d] * p;
      }
#pragma unroll
      for (int d = 0; d < 16; ++d) red[tid * 16 + d] = acc[d];
      __syncthreads();
      for (int s = RT / 2; s > 0; s >>= 1) {
        if (tid < s)
#pragma unroll
          for (int d = 0; d < 16; ++d) red[tid * 16 + d] += red[(tid + s) * 16 + d];
        __syncthreads();
      }
      if (tid == 0) {
        float ss = 0.f;
        for (int d = 0; d < 16; ++d) ss += red[d] * red[d];
        float sc = sqrtf(ss) / (1.f + ss);
        for (int d = 0; d < 16; ++d) outS[c * 16 + d] = red[d] * sc;
      }
      __syncthreads();
    }

    if (it < 3) {                 // logits += <uhat, out>
      for (int idx = tid; idx < 10 * 2048; idx += RT) {
        int n = idx & 2047;
        float uh[16];
        uhat_row(uS, dw + (size_t)idx * 128, n, uh);
        int c = idx >> 11;
        float dot = 0.f;
#pragma unroll
        for (int d = 0; d < 16; ++d) dot += uh[d] * outS[c * 16 + d];
        L[idx] += dot;
      }
      __syncthreads();
    }
  }
  for (int i = tid; i < 160; i += RT) out[b * 160 + i] = outS[i];
}

// ---------------------------------------------------------------------------
extern "C" void kernel_launch(void* const* d_in, const int* in_sizes, int n_in,
                              void* d_out, int out_size, void* d_ws, size_t ws_size,
                              hipStream_t stream) {
  const float* x      = (const float*)d_in[0];   // [128,3,32,32]
  const float* conv_w = (const float*)d_in[1];   // [256,3,9,9]
  const float* conv_b = (const float*)d_in[2];   // [256]
  const float* prim_w = (const float*)d_in[3];   // [32,8,256,9,9] == [256,256,81]
  const float* prim_b = (const float*)d_in[4];   // [32,8] == [256]
  const float* dig_w  = (const float*)d_in[5];   // [10,2048,8,16]
  float* out = (float*)d_out;                    // [128,10,16]

  char* ws = (char*)d_ws;
  size_t off = 0;
  auto carve = [&](size_t bytes) {
    void* p = ws + off;
    off += (bytes + 255) & ~(size_t)255;
    return p;
  };
  _Float16* xh   = (_Float16*)carve((size_t)BATCH * XIMG * 2);       // 1 MB
  _Float16* wt1  = (_Float16*)carve((size_t)256 * 256 * 2);          // 128 KB
  _Float16* wt2  = (_Float16*)carve((size_t)256 * K2 * 2);           // 10.6 MB
  _Float16* hbuf = (_Float16*)carve((size_t)BATCH * 256 * 576 * 2);  // 37.7 MB
  float*    u    = (float*)carve((size_t)BATCH * 2048 * 8 * 4);      // 8.4 MB
  int*      kof1 = (int*)carve((size_t)256 * 4);                     // 1 KB
  int*      kof2 = (int*)carve((size_t)K2 * 4);                      // 83 KB
  (void)in_sizes; (void)n_in; (void)out_size; (void)ws_size;

  build_koff1<<<1, 256, 0, stream>>>(kof1);
  build_koff2<<<K2 / 256, 256, 0, stream>>>(kof2);
  cvt_x<<<BATCH * XIMG / 256, 256, 0, stream>>>(x, xh);
  cvt_w1<<<256, 256, 0, stream>>>(conv_w, wt1);
  cvt_w2<<<20736, 256, 0, stream>>>(prim_w, wt2);
  conv1_wmma<<<dim3(M1 / 128, 2), 256, 0, stream>>>(xh, wt1, kof1, conv_b, hbuf);
  zero_u<<<2048, 256, 0, stream>>>((float4*)u);
  conv2_wmma<<<dim3(M2 / 128, 2, 2), 256, 0, stream>>>(hbuf, wt2, kof2, prim_b, u);
  squash_u<<<1024, 256, 0, stream>>>(u);
  const size_t smem = (size_t)(2048 * 8 + 10 * 2048 + 2048 + 2048 + RT * 16 + 160)
                      * sizeof(float);           // ~193 KB dynamic LDS
  routing_kernel<<<BATCH, RT, smem, stream>>>(u, dig_w, out);
}
